// RouteGNN_4544075399546
// MI455X (gfx1250) — matrix operand order repaired and checked by reading, not compiled, not run on hardware
//
#include <hip/hip_runtime.h>

typedef _Float16 f16;
typedef __attribute__((ext_vector_type(16))) _Float16 v16h;
typedef __attribute__((ext_vector_type(8)))  float    v8f;

__device__ inline v8f wmma16x16x32(v16h a, v16h b, v8f c) {
  return __builtin_amdgcn_wmma_f32_16x16x32_f16(false, a, false, b, (short)0, c,
                                                false, false);
}

// Pack an A-fragment (16xK tile, row-major f16 in LDS) for one 32-wide k-step.
// ISA 16-bit A layout: lanes 0-15 -> M=lane, K in {0..7, 16..23};
// lanes 16-31 -> M=lane-16, K in {8..15, 24..31}.
// j=0..7 and j=8..15 each hit 8 contiguous f16 -> compiler emits 2x ds_load_b128.
__device__ inline v16h pack_a_lds(const f16* tile, int stride, int kbase) {
  int lane = threadIdx.x & 31;
  int m = lane & 15;
  int khalf = (lane >> 4) * 8;
  v16h a;
#pragma unroll
  for (int j = 0; j < 16; ++j) {
    int k = kbase + khalf + j + ((j >= 8) ? 8 : 0);
    a[j] = tile[m * stride + k];
  }
  return a;
}

// Same but from an fp32 LDS tile (used for async-staged edge_attr).
__device__ inline v16h pack_a_lds_f32(const float* tile, int stride, int kbase) {
  int lane = threadIdx.x & 31;
  int m = lane & 15;
  int khalf = (lane >> 4) * 8;
  v16h a;
#pragma unroll
  for (int j = 0; j < 16; ++j) {
    int k = kbase + khalf + j + ((j >= 8) ? 8 : 0);
    a[j] = (f16)tile[m * stride + k];
  }
  return a;
}

// Async global->LDS copy, 16B per lane. INST_OFFSET applies to BOTH addresses.
#define ASYNC_B128(la, ga, OFF)                                              \
  asm volatile("global_load_async_to_lds_b128 %0, %1, off offset:" #OFF      \
               :: "v"(la), "v"(ga) : "memory")

// ---------------- weight fragment packing ----------------
// B-fragment layout: lane n in [0,16) holds column N=n, K in {0..7, 16..23};
// lanes 16-31 hold same column, K in {8..15, 24..31}. frag[lane*16+j], 512 f16 per tile.
__device__ void pack_one(const float* W, int K, int Ncol, f16* dst, int tid,
                         int nthr) {
  int ktiles = K >> 5, ntiles = Ncol >> 4;
  int total = ktiles * ntiles * 512;
  for (int i = tid; i < total; i += nthr) {
    int frag = i >> 9;
    int r = i & 511;
    int lane = r >> 4, j = r & 15;
    int kt = frag / ntiles, nt = frag - kt * ntiles;
    int k = kt * 32 + ((lane >> 4) << 3) + j + ((j >= 8) ? 8 : 0);
    int n = nt * 16 + (lane & 15);
    dst[frag * 512 + r] = (f16)W[(size_t)k * Ncol + n];
  }
}

__global__ void pack_weights_kernel(const float* Wn, const float* We,
                                    const float* Wl0, const float* Wr0,
                                    const float* Wl1, const float* Wr1,
                                    const float* W1, const float* W2,
                                    f16* frag) {
  int tid = blockIdx.x * blockDim.x + threadIdx.x;
  int nthr = gridDim.x * blockDim.x;
  pack_one(Wn,  128, 64, frag + 0,     tid, nthr);  // 16 frags
  pack_one(We,   32, 64, frag + 8192,  tid, nthr);  // 4
  pack_one(Wl0,  64, 64, frag + 10240, tid, nthr);  // 8
  pack_one(Wr0,  64, 64, frag + 14336, tid, nthr);  // 8
  pack_one(Wl1,  64, 64, frag + 18432, tid, nthr);  // 8
  pack_one(Wr1,  64, 64, frag + 22528, tid, nthr);  // 8
  pack_one(W1,  192, 64, frag + 26624, tid, nthr);  // 24
  pack_one(W2,   64, 64, frag + 38912, tid, nthr);  // 8
}

// ---------------- node encoder: h = relu(x @ Wn + bn) ----------------
__global__ void node_encode_kernel(const float* __restrict__ x,
                                   const f16* __restrict__ WnF,
                                   const float* __restrict__ bn,
                                   float* __restrict__ h) {
  __shared__ f16 xs[16 * 128];
  int n0 = blockIdx.x * 16;
  int tid = threadIdx.x;
  for (int i = tid; i < 16 * 128; i += 128) {
    int m = i >> 7, k = i & 127;
    xs[i] = (f16)x[(size_t)(n0 + m) * 128 + k];
  }
  __syncthreads();
  int lane = tid & 31, nt = tid >> 5;
  v8f c = {};
#pragma unroll
  for (int kt = 0; kt < 4; ++kt) {
    v16h a = pack_a_lds(xs, 128, kt * 32);
    v16h b = *(const v16h*)(WnF + (kt * 4 + nt) * 512 + lane * 16);
    c = wmma16x16x32(a, b, c);
  }
  int n = lane & 15, mhi = (lane >> 4) * 8;
  float bias = bn[nt * 16 + n];
#pragma unroll
  for (int r = 0; r < 8; ++r) {
    float v = c[r] + bias;
    h[(size_t)(n0 + r + mhi) * 64 + nt * 16 + n] = v > 0.f ? v : 0.f;
  }
}

// ---------------- degree / scatter ----------------
__global__ void degree_kernel(const int* __restrict__ dst,
                              float* __restrict__ cnt, int E) {
  int i = blockIdx.x * blockDim.x + threadIdx.x;
  if (i < E) atomicAdd(&cnt[dst[i]], 1.0f);
}

__global__ void invden_kernel(const float* __restrict__ cnt,
                              float* __restrict__ invden, int N) {
  int i = blockIdx.x * blockDim.x + threadIdx.x;
  if (i < N) invden[i] = 1.0f / fmaxf(cnt[i], 1.0f);
}

__global__ void scatter_kernel(const float* __restrict__ h,
                               const int* __restrict__ src,
                               const int* __restrict__ dst,
                               float* __restrict__ agg, int E) {
  int i = blockIdx.x * blockDim.x + threadIdx.x;
  int total = E << 6;
  if (i < total) {
    int e = i >> 6, f = i & 63;
    atomicAdd(&agg[(size_t)dst[e] * 64 + f], h[(size_t)src[e] * 64 + f]);
  }
}

// ---------------- SAGE: h' = relu(agg*invden @ Wl + bl + h @ Wr) -------------
// Optionally mirrors the result to an f16 table (used by the edge kernel).
__global__ void sage_kernel(const float* __restrict__ hin,
                            const float* __restrict__ agg,
                            const float* __restrict__ invden,
                            const f16* __restrict__ WlF,
                            const float* __restrict__ bl,
                            const f16* __restrict__ WrF,
                            float* __restrict__ hout,
                            f16* __restrict__ hout16) {
  __shared__ f16 at[16 * 64];
  __shared__ f16 ht[16 * 64];
  int n0 = blockIdx.x * 16;
  int tid = threadIdx.x;
  for (int i = tid; i < 16 * 64; i += 128) {
    int m = i >> 6, k = i & 63;
    float s = invden[n0 + m];
    at[i] = (f16)(agg[(size_t)(n0 + m) * 64 + k] * s);
    ht[i] = (f16)hin[(size_t)(n0 + m) * 64 + k];
  }
  __syncthreads();
  int lane = tid & 31, nt = tid >> 5;
  v8f c = {};
#pragma unroll
  for (int kt = 0; kt < 2; ++kt) {
    v16h a = pack_a_lds(at, 64, kt * 32);
    v16h b = *(const v16h*)(WlF + (kt * 4 + nt) * 512 + lane * 16);
    c = wmma16x16x32(a, b, c);
    v16h a2 = pack_a_lds(ht, 64, kt * 32);
    v16h b2 = *(const v16h*)(WrF + (kt * 4 + nt) * 512 + lane * 16);
    c = wmma16x16x32(a2, b2, c);
  }
  int n = lane & 15, mhi = (lane >> 4) * 8;
  float bias = bl[nt * 16 + n];
#pragma unroll
  for (int r = 0; r < 8; ++r) {
    float v = c[r] + bias;
    v = v > 0.f ? v : 0.f;
    size_t idx = (size_t)(n0 + r + mhi) * 64 + nt * 16 + n;
    hout[idx] = v;
    if (hout16) hout16[idx] = (f16)v;
  }
}

// ---------------- fused edge MLP (dominant kernel, ~59 GFLOP) ----------------
// 32 edges per block; 4 waves; wave nt owns output columns [nt*16, nt*16+16)
// and loops two 16-row M-tiles, reusing register-resident B fragments.
__global__ void edge_kernel(const f16* __restrict__ hF,
                            const int* __restrict__ srcA,
                            const int* __restrict__ dstA,
                            const float* __restrict__ eattr,
                            const f16* __restrict__ WeF, const float* __restrict__ be,
                            const f16* __restrict__ W1F, const float* __restrict__ b1,
                            const f16* __restrict__ W2F, const float* __restrict__ b2,
                            const float* __restrict__ Ws, const float* __restrict__ bs,
                            float* __restrict__ out) {
  __shared__ float et32[32 * 32];   // async-staged edge_attr tile (fp32, 4KB)
  __shared__ f16 cat[32 * 192];     // [h_src | h_dst | e_in]
  __shared__ f16 zt[32 * 64];
  __shared__ float outbuf[32];
  __shared__ int sidx[32], didx[32];
  int e0 = blockIdx.x * 32;
  int tid = threadIdx.x;

  if (tid < 32) {
    // wave 0: async copy of the contiguous 4KB edge_attr tile into LDS
    unsigned la = (unsigned)(unsigned long long)(&et32[0]) + (tid << 4);
    const char* ga = (const char*)eattr + (size_t)e0 * 128 + (tid << 4);
    ASYNC_B128(la, ga, 0);
    ASYNC_B128(la, ga, 512);
    ASYNC_B128(la, ga, 1024);
    ASYNC_B128(la, ga, 1536);
    ASYNC_B128(la, ga, 2048);
    ASYNC_B128(la, ga, 2560);
    ASYNC_B128(la, ga, 3072);
    ASYNC_B128(la, ga, 3584);
    asm volatile("s_wait_asynccnt 0x0" ::: "memory");
  } else if (tid < 64) {
    sidx[tid - 32] = srcA[e0 + tid - 32];
  } else if (tid < 96) {
    didx[tid - 64] = dstA[e0 + tid - 64];
  } else {
    outbuf[tid - 96] = bs[0];
  }
  __syncthreads();

  int lane = tid & 31, nt = tid >> 5;
  int n = lane & 15, mhi = (lane >> 4) * 8;

  // gather endpoint features from the f16 node table (16B chunks, L2-resident)
  for (int i = tid; i < 32 * 2 * 8; i += 128) {
    int m = i >> 4, half = (i >> 3) & 1, ch = i & 7;
    int node = half ? didx[m] : sidx[m];
    *((uint4*)(cat + m * 192 + half * 64) + ch) =
        *((const uint4*)(hF + (size_t)node * 64) + ch);
  }
  // e_in = edge_attr @ We + be  (no activation)
  {
    v16h b = *(const v16h*)(WeF + nt * 512 + lane * 16);
    float bias = be[nt * 16 + n];
#pragma unroll
    for (int mt = 0; mt < 2; ++mt) {
      v8f c = {};
      v16h a = pack_a_lds_f32(et32 + mt * 16 * 32, 32, 0);
      c = wmma16x16x32(a, b, c);
#pragma unroll
      for (int r = 0; r < 8; ++r)
        cat[(mt * 16 + r + mhi) * 192 + 128 + nt * 16 + n] = (f16)(c[r] + bias);
    }
  }
  __syncthreads();
  // z = relu(cat @ W1 + b1)   (B fragments kept in registers across M-tiles)
  {
    v16h bw[6];
#pragma unroll
    for (int kt = 0; kt < 6; ++kt)
      bw[kt] = *(const v16h*)(W1F + (kt * 4 + nt) * 512 + lane * 16);
    float bias = b1[nt * 16 + n];
#pragma unroll
    for (int mt = 0; mt < 2; ++mt) {
      v8f c = {};
#pragma unroll
      for (int kt = 0; kt < 6; ++kt) {
        v16h a = pack_a_lds(cat + mt * 16 * 192, 192, kt * 32);
        c = wmma16x16x32(a, bw[kt], c);
      }
#pragma unroll
      for (int r = 0; r < 8; ++r) {
        float v = c[r] + bias;
        zt[(mt * 16 + r + mhi) * 64 + nt * 16 + n] = (f16)(v > 0.f ? v : 0.f);
      }
    }
  }
  __syncthreads();
  // emb = z @ W2 + b2 ; out = emb @ Ws + bs (ds_add_f32 reduction per edge row)
  {
    v16h bw[2];
#pragma unroll
    for (int kt = 0; kt < 2; ++kt)
      bw[kt] = *(const v16h*)(W2F + (kt * 4 + nt) * 512 + lane * 16);
    float bias = b2[nt * 16 + n];
    float wsv = Ws[nt * 16 + n];
#pragma unroll
    for (int mt = 0; mt < 2; ++mt) {
      v8f c = {};
#pragma unroll
      for (int kt = 0; kt < 2; ++kt) {
        v16h a = pack_a_lds(zt + mt * 16 * 64, 64, kt * 32);
        c = wmma16x16x32(a, bw[kt], c);
      }
#pragma unroll
      for (int r = 0; r < 8; ++r)
        atomicAdd(&outbuf[mt * 16 + r + mhi], (c[r] + bias) * wsv);
    }
  }
  __syncthreads();
  if (tid < 32) out[e0 + tid] = outbuf[tid];
}

extern "C" void kernel_launch(void* const* d_in, const int* in_sizes, int n_in,
                              void* d_out, int out_size, void* d_ws, size_t ws_size,
                              hipStream_t stream) {
  const float* x     = (const float*)d_in[0];
  const int*   eidx  = (const int*)d_in[1];
  const float* eattr = (const float*)d_in[2];
  const float* Wn  = (const float*)d_in[3];
  const float* bn  = (const float*)d_in[4];
  const float* We  = (const float*)d_in[5];
  const float* be  = (const float*)d_in[6];
  const float* Wl0 = (const float*)d_in[7];
  const float* bl0 = (const float*)d_in[8];
  const float* Wr0 = (const float*)d_in[9];
  const float* Wl1 = (const float*)d_in[10];
  const float* bl1 = (const float*)d_in[11];
  const float* Wr1 = (const float*)d_in[12];
  const float* W1  = (const float*)d_in[13];
  const float* b1  = (const float*)d_in[14];
  const float* W2  = (const float*)d_in[15];
  const float* b2  = (const float*)d_in[16];
  const float* Ws  = (const float*)d_in[17];
  const float* bs  = (const float*)d_in[18];
  float* out = (float*)d_out;

  int N = in_sizes[0] / 128;
  int E = in_sizes[2] / 32;
  const int* src = eidx;
  const int* dst = eidx + E;

  char* ws = (char*)d_ws;
  size_t NH = (size_t)N * 64 * sizeof(float);
  float* hA     = (float*)ws;
  float* hB     = (float*)(ws + NH);
  float* agg    = (float*)(ws + 2 * NH);
  float* cnt    = (float*)(ws + 3 * NH);
  float* invden = (float*)(ws + 3 * NH + (size_t)N * 4);
  f16*   hF     = (f16*)(ws + 3 * NH + 2 * (size_t)N * 4);   // N*64 f16
  size_t fragOff =
      (3 * NH + 2 * (size_t)N * 4 + (size_t)N * 128 + 255) & ~(size_t)255;
  f16* frag = (f16*)(ws + fragOff);
  f16* WnF  = frag;
  f16* WeF  = frag + 8192;
  f16* Wl0F = frag + 10240;
  f16* Wr0F = frag + 14336;
  f16* Wl1F = frag + 18432;
  f16* Wr1F = frag + 22528;
  f16* W1F  = frag + 26624;
  f16* W2F  = frag + 38912;

  hipMemsetAsync(cnt, 0, (size_t)N * 4, stream);
  pack_weights_kernel<<<64, 256, 0, stream>>>(Wn, We, Wl0, Wr0, Wl1, Wr1, W1, W2, frag);
  node_encode_kernel<<<N / 16, 128, 0, stream>>>(x, WnF, bn, hA);
  degree_kernel<<<(E + 255) / 256, 256, 0, stream>>>(dst, cnt, E);
  invden_kernel<<<(N + 255) / 256, 256, 0, stream>>>(cnt, invden, N);

  hipMemsetAsync(agg, 0, NH, stream);
  scatter_kernel<<<(E * 64 + 255) / 256, 256, 0, stream>>>(hA, src, dst, agg, E);
  sage_kernel<<<N / 16, 128, 0, stream>>>(hA, agg, invden, Wl0F, bl0, Wr0F, hB,
                                          (f16*)nullptr);

  hipMemsetAsync(agg, 0, NH, stream);
  scatter_kernel<<<(E * 64 + 255) / 256, 256, 0, stream>>>(hB, src, dst, agg, E);
  sage_kernel<<<N / 16, 128, 0, stream>>>(hB, agg, invden, Wl1F, bl1, Wr1F, hA, hF);

  edge_kernel<<<E / 32, 128, 0, stream>>>(hF, src, dst, eattr, WeF, be,
                                          W1F, b1, W2F, b2, Ws, bs, out);
}
